// BWGNN_Hetero_45414984188199
// MI455X (gfx1250) — matrix operand order repaired and checked by reading, not compiled, not run on hardware
//
#include <hip/hip_runtime.h>
#include <hip/hip_bf16.h>
#include <math.h>
#include <stdint.h>

// ---------------- problem constants (from reference) ----------------
#define NN   50000
#define EE   800000
#define RRR  3
#define TTT  16
#define IVV  64     // LSTM input/hidden dim (voc & sms)
#define IPP  32     // personal feature dim
#define HHH  128

typedef __bf16 bf16;
typedef bf16  v16bf  __attribute__((ext_vector_type(16)));
typedef bf16  bf16x8 __attribute__((ext_vector_type(8)));
typedef float v8f    __attribute__((ext_vector_type(8)));
typedef unsigned u32x4 __attribute__((ext_vector_type(4)));
typedef unsigned u32x8 __attribute__((ext_vector_type(8)));

__device__ __forceinline__ bf16 f2bf(float f) {
  union { float f; unsigned u; } x; x.f = f;
  unsigned r = x.u + 0x7fffu + ((x.u >> 16) & 1u);   // round-to-nearest-even
  unsigned short h = (unsigned short)(r >> 16);
  return __builtin_bit_cast(bf16, h);
}

// ---------------- WMMA fragment loads (CDNA5 ISA layouts) ----------------
// A (16x32 bf16): lane l<16 -> row l, K {0..7,16..23}; lane l>=16 -> row l-16, K {8..15,24..31}
__device__ __forceinline__ v16bf load_a_frag(const bf16* base, int ld) {
  int lane = threadIdx.x & 31;
  const bf16* p = base + (lane & 15) * ld + ((lane >> 4) << 3);
  bf16x8 lo = *(const bf16x8*)p;
  bf16x8 hi = *(const bf16x8*)(p + 16);
  v16bf a;
#pragma unroll
  for (int i = 0; i < 8; ++i) { a[i] = lo[i]; a[i + 8] = hi[i]; }
  return a;
}
// B (32x16 bf16) for C = A @ W^T; B column n = row n of W (K-major).
__device__ __forceinline__ v16bf load_b_frag(const bf16* base, int ld) {
  int lane = threadIdx.x & 31;
  const bf16* p = base + (lane & 15) * ld + ((lane >> 4) << 3);
  bf16x8 lo = *(const bf16x8*)p;
  bf16x8 hi = *(const bf16x8*)(p + 16);
  v16bf b;
#pragma unroll
  for (int i = 0; i < 8; ++i) { b[i] = lo[i]; b[i + 8] = hi[i]; }
  return b;
}

// ---------------- generic WMMA GEMM: C = act(A @ W^T + bias + Cin) ----------------
// A: MxK bf16 (lda), W: NcolsxK bf16 (ldw), Cin optional f32, C f32.
// act: 0=none, 1=leaky_relu(0.01), 2=tanh
// A tile staged with GLOBAL_LOAD_ASYNC_TO_LDS_B128 (ASYNCcnt),
// B (weight) tile staged with the Tensor Data Mover (TENSORcnt).
__global__ __launch_bounds__(256) void gemm_bias_act(
    const bf16* __restrict__ A, int lda,
    const bf16* __restrict__ W, int ldw,
    const float* __restrict__ Cin, int ldcin,
    float* __restrict__ C, int ldc,
    const float* __restrict__ bias,
    int M, int Ncols, int K, int act)
{
  __shared__ __align__(16) bf16 As[128][32];
  __shared__ __align__(16) bf16 Bs[64][32];

  int bm = blockIdx.x * 128;
  int bn = blockIdx.y * 64;
  int tid = threadIdx.x;
  int wave = tid >> 5, lane = tid & 31;

  v8f acc[4];
#pragma unroll
  for (int j = 0; j < 4; ++j)
#pragma unroll
    for (int r = 0; r < 8; ++r) acc[j][r] = 0.f;

  int arow = tid >> 1, ahalf = (tid & 1) << 4;   // A tile: 128x32, 16 bf16/thread
  unsigned lds_a = (unsigned)(uintptr_t)&As[arow][ahalf];
  unsigned lds_b = (unsigned)(uintptr_t)&Bs[0][0];

  for (int k0 = 0; k0 < K; k0 += 32) {
    // ---- stage A tile: async global -> LDS (per-lane DMA, EXEC-masked on M edge) ----
    {
      int gr = bm + arow;
      if (gr < M) {
        const bf16* gp = A + (size_t)gr * lda + k0 + ahalf;
        asm volatile("global_load_async_to_lds_b128 %0, %1, off"
                     :: "v"(lds_a), "v"(gp) : "memory");
        asm volatile("global_load_async_to_lds_b128 %0, %1, off offset:16"
                     :: "v"(lds_a), "v"(gp) : "memory");
        if (k0 + 32 < K) __builtin_prefetch(gp + 32, 0, 0);
      }
    }
    // ---- stage B tile: one TDM descriptor per workgroup (wave 0 issues) ----
    if (wave == 0) {
      unsigned long long ga =
          (unsigned long long)(uintptr_t)(W + (size_t)bn * ldw + k0);
      u32x4 g0;
      g0[0] = 1u;                                   // count=1, user descriptor
      g0[1] = lds_b;                                // lds_addr
      g0[2] = (unsigned)(ga & 0xffffffffu);         // global_addr[31:0]
      g0[3] = (unsigned)((ga >> 32) & 0x1ffffffu)   // global_addr[56:32]
              | (2u << 30);                         // type=2 ("image")
      u32x8 g1;
      g1[0] = 1u << 16;                             // data_size=1 -> 2 bytes/elt
      g1[1] = ((unsigned)K & 0xffffu) << 16;        // tensor_dim0[15:0]
      g1[2] = (((unsigned)K >> 16) & 0xffffu)       // tensor_dim0[31:16]
              | (((unsigned)Ncols & 0xffffu) << 16);// tensor_dim1[15:0]
      g1[3] = (((unsigned)Ncols >> 16) & 0xffffu)   // tensor_dim1[31:16]
              | (32u << 16);                        // tile_dim0 = 32 elements
      g1[4] = 64u;                                  // tile_dim1 = 64 rows, tile_dim2=0
      g1[5] = (unsigned)ldw;                        // tensor_dim0_stride[31:0]
      g1[6] = 0u;                                   // stride hi, dim1 stride lo
      g1[7] = 0u;
      asm volatile("tensor_load_to_lds %0, %1" :: "s"(g0), "s"(g1) : "memory");
      __builtin_amdgcn_s_wait_tensorcnt(0);
    }
    asm volatile("s_wait_asynccnt 0x0" ::: "memory");
    __syncthreads();

    v16bf a = load_a_frag(&As[wave * 16][0], 32);
#pragma unroll
    for (int j = 0; j < 4; ++j) {
      v16bf b = load_b_frag(&Bs[j * 16][0], 32);
      acc[j] = __builtin_amdgcn_wmma_f32_16x16x32_bf16(
                   false, a, false, b, (short)0, acc[j], false, false);
    }
    __syncthreads();
  }

  // epilogue: C/D layout -> VGPR r holds M = r (lanes 0..15) / r+8 (lanes 16..31)
  int lhalf = (lane >> 4) << 3;
  int nloc  = lane & 15;
#pragma unroll
  for (int j = 0; j < 4; ++j) {
    int col = bn + j * 16 + nloc;
    if (col >= Ncols) continue;
#pragma unroll
    for (int r = 0; r < 8; ++r) {
      int row = bm + wave * 16 + lhalf + r;
      if (row >= M) continue;
      float v = acc[j][r];
      if (bias) v += bias[col];
      if (Cin)  v += Cin[(size_t)row * ldcin + col];
      if (act == 1)      v = v > 0.f ? v : 0.01f * v;
      else if (act == 2) v = tanhf(v);
      C[(size_t)row * ldc + col] = v;
    }
  }
}

// ---------------- small helper kernels ----------------
__global__ void k_f32_to_bf16(const float* __restrict__ in, bf16* __restrict__ out, long n) {
  long i = blockIdx.x * (long)blockDim.x + threadIdx.x;
  long s = (long)gridDim.x * blockDim.x;
  for (; i < n; i += s) out[i] = f2bf(in[i]);
}
__global__ void k_fill_f32(float* __restrict__ p, float v, long n) {
  long i = blockIdx.x * (long)blockDim.x + threadIdx.x;
  long s = (long)gridDim.x * blockDim.x;
  for (; i < n; i += s) p[i] = v;
}
__global__ void k_fill_bf16(bf16* __restrict__ p, long n) {
  long i = blockIdx.x * (long)blockDim.x + threadIdx.x;
  long s = (long)gridDim.x * blockDim.x;
  for (; i < n; i += s) p[i] = __builtin_bit_cast(bf16, (unsigned short)0);
}
__global__ void k_add_vec(const float* __restrict__ a, const float* __restrict__ b,
                          float* __restrict__ o, int n) {
  int i = blockIdx.x * blockDim.x + threadIdx.x;
  if (i < n) o[i] = a[i] + b[i];
}

// LSTM cell: gates [N,256] (i|f|g|o blocks of 64), updates c, writes h as bf16
__global__ void k_lstm_cell(const float* __restrict__ gates, float* __restrict__ c,
                            bf16* __restrict__ hbf, int n) {
  int i = blockIdx.x * blockDim.x + threadIdx.x;
  if (i >= n) return;
  int node = i >> 6, j = i & 63;
  const float* g = gates + (size_t)node * 256;
  float gi = g[j], gf = g[64 + j], gg = g[128 + j], go = g[192 + j];
  float si = 1.f / (1.f + expf(-gi));
  float sf = 1.f / (1.f + expf(-gf));
  float so = 1.f / (1.f + expf(-go));
  float cc = sf * c[i] + si * tanhf(gg);
  c[i] = cc;
  hbf[i] = f2bf(so * tanhf(cc));
}

// degree + dinv
__global__ void k_deg(const int* __restrict__ dst, float* __restrict__ deg, int E) {
  int e = blockIdx.x * blockDim.x + threadIdx.x;
  if (e < E) unsafeAtomicAdd(&deg[dst[e]], 1.f);
}
__global__ void k_dinv(float* __restrict__ d, int n) {
  int i = blockIdx.x * blockDim.x + threadIdx.x;
  if (i < n) { float v = d[i]; v = v < 1.f ? 1.f : v; d[i] = rsqrtf(v); }
}

// polyconv pieces (feature dim 128)
__global__ void k_pc_init(const float* __restrict__ xin, int ldx,
                          float* __restrict__ f, float* __restrict__ hfilt,
                          float th0, int fi, long n) {
  long i = blockIdx.x * (long)blockDim.x + threadIdx.x;
  if (i >= n) return;
  long node = i >> 7; int cidx = (int)(i & 127);
  float x = xin[node * ldx + cidx];
  f[i] = x;
  hfilt[(node * 5 + fi) * 128 + cidx] = th0 * x;
}
__global__ void k_scale(const float* __restrict__ f, const float* __restrict__ dinv,
                        float* __restrict__ m, long n) {
  long i = blockIdx.x * (long)blockDim.x + threadIdx.x;
  if (i >= n) return;
  m[i] = f[i] * dinv[i >> 7];
}
__global__ void k_edge_scatter(const int* __restrict__ s, const int* __restrict__ d,
                               const float* __restrict__ m, float* __restrict__ agg, long tot) {
  long tid = blockIdx.x * (long)blockDim.x + threadIdx.x;
  if (tid >= tot) return;
  long e = tid >> 5; int c4 = (int)(tid & 31) << 2;
  int sn = s[e], dn = d[e];
  float4 mv = *(const float4*)(m + (size_t)sn * 128 + c4);
  float* ap = agg + (size_t)dn * 128 + c4;
  unsafeAtomicAdd(ap + 0, mv.x);
  unsafeAtomicAdd(ap + 1, mv.y);
  unsafeAtomicAdd(ap + 2, mv.z);
  unsafeAtomicAdd(ap + 3, mv.w);
}
__global__ void k_pc_update(float* __restrict__ f, const float* __restrict__ agg,
                            const float* __restrict__ dinv, float* __restrict__ hfilt,
                            float thk, int fi, long n) {
  long i = blockIdx.x * (long)blockDim.x + threadIdx.x;
  if (i >= n) return;
  long node = i >> 7; int cidx = (int)(i & 127);
  float fn = f[i] - agg[i] * dinv[node];
  f[i] = fn;
  hfilt[(node * 5 + fi) * 128 + cidx] += thk * fn;
}

// attention over 5 filters: score=tanhed.Wf2, softmax, weighted sum of hfilt -> res
__global__ void k_attn(const float* __restrict__ tanhed, const float* __restrict__ wf2,
                       const float* __restrict__ hfilt, float* __restrict__ res, int N) {
  int wave = threadIdx.x >> 5, lane = threadIdx.x & 31;
  int n = blockIdx.x * 8 + wave;
  if (n >= N) return;
  float sc[5];
#pragma unroll
  for (int fi = 0; fi < 5; ++fi) {
    const float* tp = tanhed + ((size_t)n * 5 + fi) * 128;
    float p = 0.f;
    for (int k = lane; k < 128; k += 32) p += tp[k] * wf2[k];
#pragma unroll
    for (int o = 16; o > 0; o >>= 1) p += __shfl_xor(p, o, 32);
    sc[fi] = p;
  }
  float mx = sc[0];
#pragma unroll
  for (int fi = 1; fi < 5; ++fi) mx = fmaxf(mx, sc[fi]);
  float se = 0.f;
#pragma unroll
  for (int fi = 0; fi < 5; ++fi) { sc[fi] = expf(sc[fi] - mx); se += sc[fi]; }
  float inv = 1.f / se;
  for (int k = lane; k < 128; k += 32) {
    float v = 0.f;
#pragma unroll
    for (int fi = 0; fi < 5; ++fi)
      v += sc[fi] * inv * hfilt[((size_t)n * 5 + fi) * 128 + k];
    res[(size_t)n * 128 + k] = v;
  }
}

// final projection: out[N,2] = hfinal[N,768] @ W6[2,768]^T + b6
__global__ void k_final(const float* __restrict__ hfin, const float* __restrict__ W6,
                        const float* __restrict__ b6, float* __restrict__ out, int N) {
  int wave = threadIdx.x >> 5, lane = threadIdx.x & 31;
  int n = blockIdx.x * 8 + wave;
  if (n >= N) return;
  const float* hp = hfin + (size_t)n * 768;
  float a0 = 0.f, a1 = 0.f;
  for (int k = lane; k < 768; k += 32) {
    float h = hp[k];
    a0 += h * W6[k];
    a1 += h * W6[768 + k];
  }
#pragma unroll
  for (int o = 16; o > 0; o >>= 1) { a0 += __shfl_xor(a0, o, 32); a1 += __shfl_xor(a1, o, 32); }
  if (lane == 0) { out[(size_t)n * 2 + 0] = a0 + b6[0]; out[(size_t)n * 2 + 1] = a1 + b6[1]; }
}

// ---------------- host orchestration ----------------
extern "C" void kernel_launch(void* const* d_in, const int* in_sizes, int n_in,
                              void* d_out, int out_size, void* d_ws, size_t ws_size,
                              hipStream_t stream) {
  const float* voc   = (const float*)d_in[0];
  const float* sms   = (const float*)d_in[1];
  const float* pers  = (const float*)d_in[2];
  const float* Wih_v = (const float*)d_in[3];
  const float* Whh_v = (const float*)d_in[4];
  const float* bih_v = (const float*)d_in[5];
  const float* bhh_v = (const float*)d_in[6];
  const float* Wih_s = (const float*)d_in[7];
  const float* Whh_s = (const float*)d_in[8];
  const float* bih_s = (const float*)d_in[9];
  const float* bhh_s = (const float*)d_in[10];
  const float* W_lin  = (const float*)d_in[11];
  const float* b_lin  = (const float*)d_in[12];
  const float* W_lin1 = (const float*)d_in[13];
  const float* b_lin1 = (const float*)d_in[14];
  const float* W_pers = (const float*)d_in[15];
  const float* b_pers = (const float*)d_in[16];
  const float* W_lin2 = (const float*)d_in[17];
  const float* b_lin2 = (const float*)d_in[18];
  const float* W_lin3 = (const float*)d_in[19];
  const float* b_lin3 = (const float*)d_in[20];
  const float* W_lin4 = (const float*)d_in[21];
  const float* b_lin4 = (const float*)d_in[22];
  const float* W_lin5 = (const float*)d_in[23];
  const float* b_lin5 = (const float*)d_in[24];
  const float* Wf1    = (const float*)d_in[25];
  const float* bf1    = (const float*)d_in[26];
  const float* Wf2    = (const float*)d_in[27];
  const float* W_lin6 = (const float*)d_in[28];
  const float* b_lin6 = (const float*)d_in[29];
  const int*   src    = (const int*)d_in[30];
  const int*   dst    = (const int*)d_in[31];
  float* out = (float*)d_out;

  // workspace carve-up (256B aligned blocks)
  char* wsp = (char*)d_ws;
  auto alloc = [&](size_t bytes) -> void* {
    void* p = (void*)wsp;
    wsp += (bytes + 255) & ~(size_t)255;
    return p;
  };
  bf16* vocbf  = (bf16*)alloc((size_t)NN * TTT * IVV * 2);
  bf16* smsbf  = (bf16*)alloc((size_t)NN * TTT * IVV * 2);
  bf16* persbf = (bf16*)alloc((size_t)NN * IPP * 2);
  bf16* Wihv_bf = (bf16*)alloc((size_t)256 * 64 * 2);
  bf16* Whhv_bf = (bf16*)alloc((size_t)256 * 64 * 2);
  bf16* Wihs_bf = (bf16*)alloc((size_t)256 * 64 * 2);
  bf16* Whhs_bf = (bf16*)alloc((size_t)256 * 64 * 2);
  bf16* Wlin_bf  = (bf16*)alloc((size_t)128 * 64 * 2);
  bf16* Wlin1_bf = (bf16*)alloc((size_t)128 * 64 * 2);
  bf16* Wpers_bf = (bf16*)alloc((size_t)128 * 32 * 2);
  bf16* Wlin2_bf = (bf16*)alloc((size_t)128 * 256 * 2);
  bf16* Wlin3_bf = (bf16*)alloc((size_t)128 * 256 * 2);
  bf16* Wlin4_bf = (bf16*)alloc((size_t)128 * 512 * 2);
  bf16* Wlin5_bf = (bf16*)alloc((size_t)RRR * 128 * 128 * 2);
  bf16* Wf1_bf   = (bf16*)alloc((size_t)RRR * 128 * 128 * 2);
  float* bsum_v = (float*)alloc(256 * 4);
  float* bsum_s = (float*)alloc(256 * 4);
  float* gates  = (float*)alloc((size_t)NN * 256 * 4);
  float* cbuf   = (float*)alloc((size_t)NN * 64 * 4);
  bf16*  hbuf   = (bf16*)alloc((size_t)NN * 64 * 2);
  float* xvs    = (float*)alloc((size_t)NN * 512 * 4);
  bf16*  xvs_bf = (bf16*)alloc((size_t)NN * 512 * 2);
  float* hfinal = (float*)alloc((size_t)NN * 768 * 4);
  float* dinv   = (float*)alloc((size_t)NN * 4);
  float* ffeat  = (float*)alloc((size_t)NN * 128 * 4);
  float* mbuf   = (float*)alloc((size_t)NN * 128 * 4);
  float* agg    = (float*)alloc((size_t)NN * 128 * 4);
  float* hfilt  = (float*)alloc((size_t)NN * 5 * 128 * 4);
  bf16*  hfilt_bf = (bf16*)alloc((size_t)NN * 5 * 128 * 2);
  float* tanhed = (float*)alloc((size_t)NN * 5 * 128 * 4);
  float* res    = (float*)alloc((size_t)NN * 128 * 4);
  bf16*  res_bf = (bf16*)alloc((size_t)NN * 128 * 2);

  auto cdiv = [](long a, long b) -> unsigned { return (unsigned)((a + b - 1) / b); };
  auto conv = [&](const float* in, bf16* o, long n) {
    long blk = (n + 255) / 256; if (blk > 65535) blk = 65535;
    k_f32_to_bf16<<<dim3((unsigned)blk), dim3(256), 0, stream>>>(in, o, n);
  };
  auto fillf = [&](float* p, float v, long n) {
    long blk = (n + 255) / 256; if (blk > 65535) blk = 65535;
    k_fill_f32<<<dim3((unsigned)blk), dim3(256), 0, stream>>>(p, v, n);
  };
  auto fillb = [&](bf16* p, long n) {
    long blk = (n + 255) / 256; if (blk > 65535) blk = 65535;
    k_fill_bf16<<<dim3((unsigned)blk), dim3(256), 0, stream>>>(p, n);
  };
  auto gemm = [&](const bf16* A, int lda, const bf16* W, int ldw,
                  const float* Cin, int ldcin, float* C, int ldc,
                  const float* bias, int M, int Nc, int K, int act) {
    dim3 g(cdiv(M, 128), cdiv(Nc, 64));
    gemm_bias_act<<<g, dim3(256), 0, stream>>>(A, lda, W, ldw, Cin, ldcin, C, ldc,
                                               bias, M, Nc, K, act);
  };

  // ---- convert inputs/weights to bf16 ----
  conv(voc, vocbf, (long)NN * TTT * IVV);
  conv(sms, smsbf, (long)NN * TTT * IVV);
  conv(pers, persbf, (long)NN * IPP);
  conv(Wih_v, Wihv_bf, 256 * 64);  conv(Whh_v, Whhv_bf, 256 * 64);
  conv(Wih_s, Wihs_bf, 256 * 64);  conv(Whh_s, Whhs_bf, 256 * 64);
  conv(W_lin,  Wlin_bf, 128 * 64); conv(W_lin1, Wlin1_bf, 128 * 64);
  conv(W_pers, Wpers_bf, 128 * 32);
  conv(W_lin2, Wlin2_bf, 128 * 256); conv(W_lin3, Wlin3_bf, 128 * 256);
  conv(W_lin4, Wlin4_bf, 128 * 512);
  conv(W_lin5, Wlin5_bf, (long)RRR * 128 * 128);
  conv(Wf1,    Wf1_bf,   (long)RRR * 128 * 128);
  k_add_vec<<<1, 256, 0, stream>>>(bih_v, bhh_v, bsum_v, 256);
  k_add_vec<<<1, 256, 0, stream>>>(bih_s, bhh_s, bsum_s, 256);

  // ---- LSTM runner (gates = x_t@Wih^T + bsum; gates += h@Whh^T; cell) ----
  auto run_lstm = [&](const bf16* xbf, const bf16* Wih, const bf16* Whh, const float* bsum) {
    fillf(cbuf, 0.f, (long)NN * 64);
    fillb(hbuf, (long)NN * 64);
    for (int t = 0; t < TTT; ++t) {
      gemm(xbf + (size_t)t * IVV, TTT * IVV, Wih, IVV, nullptr, 0,
           gates, 256, bsum, NN, 256, IVV, 0);
      gemm(hbuf, 64, Whh, 64, gates, 256, gates, 256, nullptr, NN, 256, 64, 0);
      k_lstm_cell<<<cdiv((long)NN * 64, 256), 256, 0, stream>>>(gates, cbuf, hbuf, NN * 64);
    }
  };

  // ---- voc LSTM -> xv (activated) into xvs[:,0:128] ----
  run_lstm(vocbf, Wihv_bf, Whhv_bf, bsum_v);
  gemm(hbuf, 64, Wlin_bf, 64, nullptr, 0, xvs + 0, 512, b_lin, NN, 128, 64, 1);
  // xp (activated) into xvs[:,128:256] and xvs[:,384:512]
  gemm(persbf, 32, Wpers_bf, 32, nullptr, 0, xvs + 128, 512, b_pers, NN, 128, 32, 1);
  gemm(persbf, 32, Wpers_bf, 32, nullptr, 0, xvs + 384, 512, b_pers, NN, 128, 32, 1);

  // ---- sms LSTM -> xs (activated) into xvs[:,256:384] ----
  run_lstm(smsbf, Wihs_bf, Whhs_bf, bsum_s);
  gemm(hbuf, 64, Wlin1_bf, 64, nullptr, 0, xvs + 256, 512, b_lin1, NN, 128, 64, 1);

  conv(xvs, xvs_bf, (long)NN * 512);

  // ---- x_in[0..2] (activated) into hfinal cols 384/512/640 ----
  gemm(xvs_bf + 0,   512, Wlin2_bf, 256, nullptr, 0, hfinal + 384, 768, b_lin2, NN, 128, 256, 1);
  gemm(xvs_bf + 256, 512, Wlin3_bf, 256, nullptr, 0, hfinal + 512, 768, b_lin3, NN, 128, 256, 1);
  gemm(xvs_bf + 0,   512, Wlin4_bf, 512, nullptr, 0, hfinal + 640, 768, b_lin4, NN, 128, 512, 1);

  // ---- graph relations ----
  static const float THV[5][3] = {{0.8f, -0.5f, 0.f}, {3.f, -3.f, 0.75f},
                                  {0.f, 3.f, -1.5f}, {0.f, 0.f, 0.75f},
                                  {-0.2f, 0.5f, 0.f}};
  static const int THL[5] = {2, 3, 3, 3, 2};

  for (int r = 0; r < RRR; ++r) {
    const int* sr = src + (size_t)r * EE;
    const int* dr = dst + (size_t)r * EE;
    fillf(dinv, 0.f, NN);
    k_deg<<<cdiv(EE, 256), 256, 0, stream>>>(dr, dinv, EE);
    k_dinv<<<cdiv(NN, 256), 256, 0, stream>>>(dinv, NN);

    const float* xin = hfinal + 384 + 128 * r;   // x_in[r], row stride 768
    long nfe = (long)NN * 128;
    for (int fi = 0; fi < 5; ++fi) {
      k_pc_init<<<cdiv(nfe, 256), 256, 0, stream>>>(xin, 768, ffeat, hfilt,
                                                    THV[fi][0], fi, nfe);
      for (int k = 1; k < THL[fi]; ++k) {
        k_scale<<<cdiv(nfe, 256), 256, 0, stream>>>(ffeat, dinv, mbuf, nfe);
        fillf(agg, 0.f, nfe);
        k_edge_scatter<<<cdiv((long)EE * 32, 256), 256, 0, stream>>>(
            sr, dr, mbuf, agg, (long)EE * 32);
        k_pc_update<<<cdiv(nfe, 256), 256, 0, stream>>>(ffeat, agg, dinv, hfilt,
                                                        THV[fi][k], fi, nfe);
      }
    }

    conv(hfilt, hfilt_bf, (long)NN * 5 * 128);
    // tanh(hfilt @ Wf1[r]^T + bf1[r]) over M = N*5 rows
    gemm(hfilt_bf, 128, Wf1_bf + (size_t)r * 128 * 128, 128, nullptr, 0,
         tanhed, 128, bf1 + (size_t)r * 128, NN * 5, 128, 128, 2);
    k_attn<<<cdiv(NN, 8), 256, 0, stream>>>(tanhed, Wf2 + (size_t)r * 128, hfilt, res, NN);
    conv(res, res_bf, (long)NN * 128);
    // leaky(res @ W_lin5[r]^T + b_lin5[r]) into hfinal cols 128*r
    gemm(res_bf, 128, Wlin5_bf + (size_t)r * 128 * 128, 128, nullptr, 0,
         hfinal + 128 * r, 768, b_lin5 + (size_t)r * 128, NN, 128, 128, 1);
  }

  // ---- final projection to [N, 2] ----
  k_final<<<cdiv(NN, 8), 256, 0, stream>>>(hfinal, W_lin6, b_lin6, out, NN);
  (void)in_sizes; (void)n_in; (void)out_size; (void)ws_size;
}
